// NJM_87814901334119
// MI455X (gfx1250) — matrix operand order repaired and checked by reading, not compiled, not run on hardware
//
#include <hip/hip_runtime.h>

// ---------------------------------------------------------------------------
// LayerNormLSTM on MI455X (gfx1250): bf16-split WMMA GEMMs with TDM-staged
// weights in LDS + fused LN cell.
// ---------------------------------------------------------------------------

typedef __attribute__((ext_vector_type(16))) __bf16 v16bf;
typedef __attribute__((ext_vector_type(8)))  __bf16 v8bf;
typedef __attribute__((ext_vector_type(8)))  float  v8f;
typedef __attribute__((ext_vector_type(4)))  unsigned int v4u;
typedef __attribute__((ext_vector_type(8)))  int v8i;
typedef __attribute__((ext_vector_type(4)))  int v4i;

#define T_STEPS 256
#define BATCH   128
#define HID     1024
#define GATES   4096   // 4*HID

// Load two contiguous 16-byte bf16 chunks and concatenate into one v16bf.
__device__ __forceinline__ v16bf ld2x8(const __bf16* p0, const __bf16* p1) {
    v8bf a = *reinterpret_cast<const v8bf*>(p0);
    v8bf b = *reinterpret_cast<const v8bf*>(p1);
    return __builtin_shufflevector(a, b, 0,1,2,3,4,5,6,7,8,9,10,11,12,13,14,15);
}

// TDM: DMA a 2-D tile (rows x kelems of 2-byte elements, row stride = stride0
// elements) from global memory into LDS at byte offset lds_off.
// D# layout per CDNA5 ISA ch.8.3/8.4 (groups 2/3 zero => 2-D tensor).
// This toolchain exposes the 6-arg builtin: (g0, g1, g2, g3, g4, cpol).
__device__ __forceinline__ void tdm_load_2d(unsigned lds_off, const void* gaddr,
                                            unsigned rows, unsigned kelems,
                                            unsigned stride0) {
    unsigned long long ga = (unsigned long long)(size_t)gaddr;
    v4u g0;
    g0[0] = 1u;                                   // count=1 (valid), user mode
    g0[1] = lds_off;                              // lds_addr (bytes)
    g0[2] = (unsigned)(ga & 0xFFFFFFFFu);         // global_addr[31:0]
    g0[3] = (unsigned)((ga >> 32) & 0x01FFFFFFu)  // global_addr[56:32]
          | (2u << 30);                           // type = 2 ("image")
    v8i g1;
    g1[0] = (int)(1u << 16);                      // wg_mask=0, data_size=1 (2B)
    g1[1] = (int)((kelems & 0xFFFFu) << 16);      // tensor_dim0[15:0]
    g1[2] = (int)(((kelems >> 16) & 0xFFFFu)      // tensor_dim0[31:16]
          | ((rows & 0xFFFFu) << 16));            // tensor_dim1[15:0]
    g1[3] = (int)(((rows >> 16) & 0xFFFFu)        // tensor_dim1[31:16]
          | ((kelems & 0xFFFFu) << 16));          // tile_dim0
    g1[4] = (int)(rows & 0xFFFFu);                // tile_dim1 (tile_dim2 = 0)
    g1[5] = (int)stride0;                         // tensor_dim0_stride[31:0]
    g1[6] = 0;                                    // stride0[47:32] | dim1_stride lo
    g1[7] = 0;
    v4i gz4  = {0, 0, 0, 0};
    v8i gz8  = {0, 0, 0, 0, 0, 0, 0, 0};
    __builtin_amdgcn_tensor_load_to_lds(g0, g1, gz4, gz4, gz8, 0);
}

// fp32 -> bf16 hi + bf16 lo residual (split precision for WMMA).
__global__ __launch_bounds__(256) void cvt_split(const float* __restrict__ src,
                                                 __bf16* __restrict__ hi,
                                                 __bf16* __restrict__ lo, int n) {
    int i = blockIdx.x * 256 + threadIdx.x;
    if (i < n) {
        float v = src[i];
        __bf16 h = (__bf16)v;
        hi[i] = h;
        lo[i] = (__bf16)(v - (float)h);
    }
}

__global__ __launch_bounds__(256) void init_state(__bf16* __restrict__ h_hi,
                                                  __bf16* __restrict__ h_lo,
                                                  float* __restrict__ c, int n) {
    int i = blockIdx.x * 256 + threadIdx.x;
    if (i < n) {
        h_hi[i] = (__bf16)0.0f;
        h_lo[i] = (__bf16)0.0f;
        c[i]    = 0.0f;
    }
}

// Dual GEMM: P[b,g] = sum_k A[b,k] * W[g,k]   (A:[128,1024], W:[4096,1024])
// blockIdx.y == 0 -> (x_t, w_ih) ; blockIdx.y == 1 -> (h, w_hh)
// Block = 256 threads = 8 waves. Wave w computes M-tile w (16 batch rows)
// over a 64-column N strip (4 WMMA tiles). The block's full B slice
// (64 cols x 1024 K, hi+lo bf16 = 256 KB) is TDM-DMA'd into LDS once,
// then the K loop reads B via ds_load_b128. bf16 hi/lo split: 3 WMMAs/tile.
__global__ __launch_bounds__(256) void gemm_dual(
    const __bf16* __restrict__ A0hi, const __bf16* __restrict__ A0lo,
    const __bf16* __restrict__ W0hi, const __bf16* __restrict__ W0lo,
    float* __restrict__ P0,
    const __bf16* __restrict__ A1hi, const __bf16* __restrict__ A1lo,
    const __bf16* __restrict__ W1hi, const __bf16* __restrict__ W1lo,
    float* __restrict__ P1) {
    extern __shared__ __align__(128) char smemB[];   // 2 * 64*1024*2 bytes

    const bool sel = (blockIdx.y != 0);
    const __bf16* Ahi = sel ? A1hi : A0hi;
    const __bf16* Alo = sel ? A1lo : A0lo;
    const __bf16* Whi = sel ? W1hi : W0hi;
    const __bf16* Wlo = sel ? W1lo : W0lo;
    float*        P   = sel ? P1   : P0;

    const int g0 = blockIdx.x * 64;                // gate-column strip base
    const unsigned HALF = 64u * HID * 2u;          // 128 KB per array

    // LDS byte offset of the dynamic shared buffer (LDS aperture: addr[31:0]).
    const unsigned ldsBase = (unsigned)(unsigned long long)(size_t)smemB;

    if (threadIdx.x == 0) {
        tdm_load_2d(ldsBase,        Whi + (size_t)g0 * HID, 64u, HID, HID);
        tdm_load_2d(ldsBase + HALF, Wlo + (size_t)g0 * HID, 64u, HID, HID);
        __builtin_amdgcn_s_wait_tensorcnt(0);
    }
    __syncthreads();

    const __bf16* lBhi = (const __bf16*)smemB;            // [64][HID]
    const __bf16* lBlo = (const __bf16*)(smemB + HALF);   // [64][HID]

    const int lane  = threadIdx.x & 31;
    const int wave  = threadIdx.x >> 5;
    const int m0    = wave * 16;           // batch-row tile base
    const int l15   = lane & 15;
    const int upper = (lane >> 4) & 1;     // lanes 16..31 hold the other K half

    // A-matrix (16x32 bf16, MxK): lane l<16 -> M=l, K {0..7}&{16..23};
    // lanes 16..31 -> same M, K {8..15}&{24..31}.
    const __bf16* aHi = Ahi + (size_t)(m0 + l15) * HID + upper * 8;
    const __bf16* aLo = Alo + (size_t)(m0 + l15) * HID + upper * 8;

    // B-matrix (32x16 bf16, KxN): lane n<16 holds column n, K {0..7}&{16..23};
    // lanes 16..31 hold K {8..15}&{24..31}. LDS row (column of W) is contiguous over K.
    const __bf16* bHi[4];
    const __bf16* bLo[4];
#pragma unroll
    for (int t = 0; t < 4; ++t) {
        const int row = (t * 16 + l15) * HID + upper * 8;
        bHi[t] = lBhi + row;
        bLo[t] = lBlo + row;
    }

    const v8f vzero = {0.f, 0.f, 0.f, 0.f, 0.f, 0.f, 0.f, 0.f};
    v8f acc[4];
#pragma unroll
    for (int t = 0; t < 4; ++t) acc[t] = vzero;

    for (int k0 = 0; k0 < HID; k0 += 32) {
        v16bf ahi = ld2x8(aHi + k0, aHi + k0 + 16);
        v16bf alo = ld2x8(aLo + k0, aLo + k0 + 16);
#pragma unroll
        for (int t = 0; t < 4; ++t) {
            v16bf whi = ld2x8(bHi[t] + k0, bHi[t] + k0 + 16);
            v16bf wlo = ld2x8(bLo[t] + k0, bLo[t] + k0 + 16);
            acc[t] = __builtin_amdgcn_wmma_f32_16x16x32_bf16(
                false, ahi, false, whi, (short)0, acc[t], false, false);
            acc[t] = __builtin_amdgcn_wmma_f32_16x16x32_bf16(
                false, ahi, false, wlo, (short)0, acc[t], false, false);
            acc[t] = __builtin_amdgcn_wmma_f32_16x16x32_bf16(
                false, alo, false, whi, (short)0, acc[t], false, false);
        }
    }

    // C/D layout: VGPR r, lanes 0..15 -> M=r, lanes 16..31 -> M=r+8; N = lane&15.
#pragma unroll
    for (int t = 0; t < 4; ++t) {
        const int col = g0 + t * 16 + l15;
#pragma unroll
        for (int r = 0; r < 8; ++r) {
            const int row = m0 + (upper ? r + 8 : r);
            P[(size_t)row * GATES + col] = acc[t][r];
        }
    }
}

__device__ __forceinline__ void blockReduce2(float& v1, float& v2,
                                             float* r1, float* r2) {
    const int t = threadIdx.x;
    r1[t] = v1; r2[t] = v2;
    __syncthreads();
    for (int off = 128; off > 0; off >>= 1) {
        if (t < off) { r1[t] += r1[t + off]; r2[t] += r2[t + off]; }
        __syncthreads();
    }
    v1 = r1[0]; v2 = r2[0];
    __syncthreads();
}

// One block per batch row: LN(ih)+LN(hh) gates, LSTM cell, LN(c), emit y/h/c.
__global__ __launch_bounds__(256) void lstm_cell(
    const float* __restrict__ Pih, const float* __restrict__ Phh,
    const float* __restrict__ b_ih, const float* __restrict__ b_hh,
    const float* __restrict__ g_ih, const float* __restrict__ be_ih,
    const float* __restrict__ g_hh, const float* __restrict__ be_hh,
    const float* __restrict__ g_ho, const float* __restrict__ be_ho,
    float* __restrict__ c, __bf16* __restrict__ h_hi, __bf16* __restrict__ h_lo,
    float* __restrict__ y_t) {
    __shared__ float sA[GATES];   // ih preacts, then combined gates
    __shared__ float sB[GATES];   // hh preacts, then [0..HID) reused for o-gate
    __shared__ float sC[HID];     // new cell state
    __shared__ float r1[256];
    __shared__ float r2[256];

    const int b   = blockIdx.x;
    const int tid = threadIdx.x;

    float si = 0.f, qi = 0.f, sh = 0.f, qh = 0.f;
    for (int g = tid; g < GATES; g += 256) {
        float ai = Pih[(size_t)b * GATES + g] + b_ih[g];
        float ah = Phh[(size_t)b * GATES + g] + b_hh[g];
        sA[g] = ai; sB[g] = ah;
        si += ai; qi += ai * ai;
        sh += ah; qh += ah * ah;
    }
    blockReduce2(si, qi, r1, r2);
    const float mu_i = si * (1.f / GATES);
    const float rs_i = rsqrtf(fmaxf(qi * (1.f / GATES) - mu_i * mu_i, 0.f) + 1e-5f);
    blockReduce2(sh, qh, r1, r2);
    const float mu_h = sh * (1.f / GATES);
    const float rs_h = rsqrtf(fmaxf(qh * (1.f / GATES) - mu_h * mu_h, 0.f) + 1e-5f);

    for (int g = tid; g < GATES; g += 256) {
        float gv = (sA[g] - mu_i) * rs_i * g_ih[g] + be_ih[g]
                 + (sB[g] - mu_h) * rs_h * g_hh[g] + be_hh[g];
        sA[g] = gv;
    }
    __syncthreads();

    float cs = 0.f, cq = 0.f;
    for (int j = tid; j < HID; j += 256) {
        float iv = 1.f / (1.f + expf(-sA[j]));
        float fv = 1.f / (1.f + expf(-sA[HID + j]));
        float ov = 1.f / (1.f + expf(-sA[2 * HID + j]));
        float gv = tanhf(sA[3 * HID + j]);
        float cn = fv * c[(size_t)b * HID + j] + iv * gv;
        sC[j] = cn; sB[j] = ov;
        cs += cn; cq += cn * cn;
    }
    blockReduce2(cs, cq, r1, r2);
    const float mu_c = cs * (1.f / HID);
    const float rs_c = rsqrtf(fmaxf(cq * (1.f / HID) - mu_c * mu_c, 0.f) + 1e-5f);

    for (int j = tid; j < HID; j += 256) {
        float cn = sC[j];
        float hv = sB[j] * tanhf((cn - mu_c) * rs_c * g_ho[j] + be_ho[j]);
        c[(size_t)b * HID + j]   = cn;
        y_t[(size_t)b * HID + j] = hv;
        __bf16 hh = (__bf16)hv;
        h_hi[(size_t)b * HID + j] = hh;
        h_lo[(size_t)b * HID + j] = (__bf16)(hv - (float)hh);
    }
}

extern "C" void kernel_launch(void* const* d_in, const int* in_sizes, int n_in,
                              void* d_out, int out_size, void* d_ws, size_t ws_size,
                              hipStream_t stream) {
    (void)in_sizes; (void)n_in; (void)out_size; (void)ws_size;

    const float* x     = (const float*)d_in[0];
    const float* w_ih  = (const float*)d_in[1];
    const float* w_hh  = (const float*)d_in[2];
    const float* b_ih  = (const float*)d_in[3];
    const float* b_hh  = (const float*)d_in[4];
    const float* g_ih  = (const float*)d_in[5];
    const float* be_ih = (const float*)d_in[6];
    const float* g_hh  = (const float*)d_in[7];
    const float* be_hh = (const float*)d_in[8];
    const float* g_ho  = (const float*)d_in[9];
    const float* be_ho = (const float*)d_in[10];
    float* y = (float*)d_out;

    char* ws = (char*)d_ws;
    size_t off = 0;
    auto take = [&](size_t bytes) -> void* {
        void* p = ws + off;
        off += (bytes + 255) & ~(size_t)255;
        return p;
    };

    const size_t WN = (size_t)GATES * HID;     // weight elements
    const size_t SN = (size_t)BATCH * HID;     // state elements

    __bf16* wih_hi = (__bf16*)take(WN * 2);
    __bf16* wih_lo = (__bf16*)take(WN * 2);
    __bf16* whh_hi = (__bf16*)take(WN * 2);
    __bf16* whh_lo = (__bf16*)take(WN * 2);
    __bf16* xt_hi  = (__bf16*)take(SN * 2);
    __bf16* xt_lo  = (__bf16*)take(SN * 2);
    __bf16* h_hi   = (__bf16*)take(SN * 2);
    __bf16* h_lo   = (__bf16*)take(SN * 2);
    float*  c_st   = (float*)take(SN * 4);
    float*  Pih    = (float*)take((size_t)BATCH * GATES * 4);
    float*  Phh    = (float*)take((size_t)BATCH * GATES * 4);

    cvt_split<<<(int)((WN + 255) / 256), 256, 0, stream>>>(w_ih, wih_hi, wih_lo, (int)WN);
    cvt_split<<<(int)((WN + 255) / 256), 256, 0, stream>>>(w_hh, whh_hi, whh_lo, (int)WN);
    init_state<<<(int)((SN + 255) / 256), 256, 0, stream>>>(h_hi, h_lo, c_st, (int)SN);

    const size_t smemBytes = 2u * 64u * HID * 2u;   // 256 KB: B hi + B lo slices

    for (int t = 0; t < T_STEPS; ++t) {
        cvt_split<<<(int)((SN + 255) / 256), 256, 0, stream>>>(
            x + (size_t)t * SN, xt_hi, xt_lo, (int)SN);
        gemm_dual<<<dim3(64, 2, 1), 256, smemBytes, stream>>>(
            xt_hi, xt_lo, wih_hi, wih_lo, Pih,
            h_hi,  h_lo,  whh_hi, whh_lo, Phh);
        lstm_cell<<<BATCH, 256, 0, stream>>>(
            Pih, Phh, b_ih, b_hh, g_ih, be_ih, g_hh, be_hh, g_ho, be_ho,
            c_st, h_hi, h_lo, y + (size_t)t * SN);
    }
}
